// w_o_FCM_28063316312600
// MI455X (gfx1250) — matrix operand order repaired and checked by reading, not compiled, not run on hardware
//
// CDNA5 / gfx1250 implementation of the 3-level PLE/CGC multi-task MoE forward.
// ~2.3 TFLOP vs ~1.3GB unique data (AI ~ 1800 FLOP/B) -> compute bound ->
// all GEMMs via v_wmma_f32_16x16x32_f16 (f16 in, f32 accumulate).
// B operand is staged per-block in LDS (double-buffered, 64-K-deep stages) with
// global_load_async_to_lds_b128 + s_wait_asynccnt (ASYNCcnt path): 4x block-level
// B reuse, global latency overlapped with the WMMA stream, and only one barrier
// pair per 8 WMMAs.

#include <hip/hip_runtime.h>
#include <hip/hip_bf16.h>
#include <math.h>

typedef _Float16 f16;
typedef __attribute__((ext_vector_type(16))) _Float16 v16h;
typedef __attribute__((ext_vector_type(8)))  _Float16 v8h;
typedef __attribute__((ext_vector_type(8)))  float    v8f;

#define BATCH  4096
#define DHID   2048
#define THID   128
#define NGATE  8     // E_SPEC + E_SH

#define BN_TILE 128          // block N tile
#define KSTEP   64           // K depth per LDS stage (all K here are multiples of 64)
#define LDSROW  72           // f16 per LDS row: 64 data + 8 pad (144B = 36 banks)

// ------------------------------------------------------------------ convert
__global__ void k_cvt_f32_f16(const float* __restrict__ src, f16* __restrict__ dst, long n) {
  long i = (long)blockIdx.x * blockDim.x + threadIdx.x;
  long stride = (long)gridDim.x * blockDim.x;
  for (; i < n; i += stride) dst[i] = (f16)src[i];
}

// transpose + downconvert: src f32 (R x C, row stride ld) -> dst f16 (C x R, row stride dstld)
__global__ void k_trans_f32_f16(const float* __restrict__ src, int R, int C, int ld,
                                f16* __restrict__ dst, int dstld) {
  __shared__ float tile[32][33];
  int c0 = blockIdx.x * 32, r0 = blockIdx.y * 32;
  int tx = threadIdx.x, ty = threadIdx.y;            // block = 32 x 8
  #pragma unroll
  for (int j = 0; j < 32; j += 8) {
    int r = r0 + ty + j, c = c0 + tx;
    tile[ty + j][tx] = (r < R && c < C) ? src[(size_t)r * ld + c] : 0.f;
  }
  __syncthreads();
  #pragma unroll
  for (int j = 0; j < 32; j += 8) {
    int c = c0 + ty + j, r = r0 + tx;
    if (c < C && r < R) dst[(size_t)c * dstld + r] = (f16)tile[tx][ty + j];
  }
}

// ------------------------------------------------------------------ WMMA GEMM
// C(MxN) = alpha * A(MxK) * B(KxN) + bias[N]; B passed transposed: Bt (N x K, ldb).
// Optional ReLU; writes f32 (Cf) and/or f16 (Ch).
// Block = 256 threads = 8 waves (4 along M x 2 along N); wave tile 16x64
// (4 wmma accumulators); block tile 64x128. M multiple of 64, K multiple of 64
// (true at every call site); N guarded (gates N=8).
// B tile (128 rows x 64 K of f16) async-staged into LDS, double buffered:
// 8 WMMAs per barrier pair.
__global__ void __launch_bounds__(256)
k_gemm_wmma(const f16* __restrict__ A, int lda,
            const f16* __restrict__ Bt, int ldb,
            const float* __restrict__ bias, float alpha,
            float* __restrict__ Cf, int ldc,
            f16* __restrict__ Ch, int ldch,
            int M, int N, int K, int relu) {
  __shared__ f16 bsh[2][BN_TILE * LDSROW];           // 2 x 18KB stages

  const int tid  = threadIdx.x;
  const int wave = tid >> 5;
  const int lane = tid & 31;
  const int lm   = lane & 15;
  const int lh   = lane >> 4;                        // K-half select
  const int wm   = wave & 3, wn = wave >> 2;
  const int m0   = blockIdx.y * 64 + wm * 16;
  const int n0b  = blockIdx.x * BN_TILE;

  const uint32_t lds0 = (uint32_t)(size_t)(&bsh[0][0]);   // flat low 32b = LDS offset

  // cooperatively async-load Bt[n0b..n0b+127][kk..kk+63] into stage s:
  // 1024 chunks of 16B, 4 per thread, row-clamped for N<128 (garbage confined
  // to unstored columns: wmma output column n only consumes B row n).
  auto stageB = [&](int s, int kk) {
    #pragma unroll
    for (int t = 0; t < 4; ++t) {
      int c   = tid + t * 256;                       // chunk 0..1023
      int row = c >> 3;
      int ko  = (c & 7) * 8;                         // f16 elems within 64-K slice
      int n   = n0b + row; n = (n < N) ? n : 0;
      const f16* gp = Bt + (size_t)n * ldb + kk + ko;
      uint32_t lo = lds0 + (uint32_t)(s * (BN_TILE * LDSROW * 2) +
                                      row * (LDSROW * 2) + ko * 2);
      asm volatile("global_load_async_to_lds_b128 %0, %1, off"
                   :: "v"(lo), "v"(gp) : "memory");
    }
  };

  v8f acc[4] = {};
  const f16* ap = A + (size_t)(m0 + lm) * lda;       // A-frag row for this lane

  stageB(0, 0);
  int s = 0;
  for (int kk = 0; kk < K; kk += KSTEP, s ^= 1) {
    if (kk + KSTEP < K) {
      stageB(s ^ 1, kk + KSTEP);                     // prefetch next stage
      asm volatile("s_wait_asynccnt 0x4" ::: "memory");   // oldest 4 (cur) done
    } else {
      asm volatile("s_wait_asynccnt 0x0" ::: "memory");
    }
    __syncthreads();                                 // stage s visible to all waves

    if (kk + KSTEP < K) __builtin_prefetch(ap + kk + KSTEP, 0, 3);
    #pragma unroll
    for (int half = 0; half < 2; ++half) {           // two 32-K sub-steps
      const int kb = half * 32;
      // A fragment: elems 0..7 = K[lh*8..], elems 8..15 = K[16+lh*8..]
      v8h alo = *(const v8h*)(ap + kk + kb + lh * 8);
      v8h ahi = *(const v8h*)(ap + kk + kb + 16 + lh * 8);
      v16h af = __builtin_shufflevector(alo, ahi,
                  0,1,2,3,4,5,6,7,8,9,10,11,12,13,14,15);
      #pragma unroll
      for (int t = 0; t < 4; ++t) {
        const f16* lp = &bsh[s][(wn * 64 + t * 16 + lm) * LDSROW + kb + lh * 8];
        v8h b0 = *(const v8h*)lp;                    // ds_load_b128
        v8h b1 = *(const v8h*)(lp + 16);
        v16h bf = __builtin_shufflevector(b0, b1,
                    0,1,2,3,4,5,6,7,8,9,10,11,12,13,14,15);
        acc[t] = __builtin_amdgcn_wmma_f32_16x16x32_f16(
            false, af, false, bf, (short)0, acc[t], false, false);
      }
    }
    __syncthreads();                                 // reads done before overwrite
  }

  #pragma unroll
  for (int t = 0; t < 4; ++t) {
    int col = n0b + wn * 64 + t * 16 + lm;
    if (col < N) {
      float bv = bias ? bias[col] : 0.f;
      #pragma unroll
      for (int r = 0; r < 8; ++r) {
        int row = m0 + lh * 8 + r;                   // C/D layout: vgpr r, half lh
        float v = alpha * acc[t][r] + bv;
        if (relu) v = fmaxf(v, 0.f);
        if (Cf) Cf[(size_t)row * ldc  + col] = v;
        if (Ch) Ch[(size_t)row * ldch + col] = (f16)v;
      }
    }
  }
}

// ------------------------------------------------------------------ softmax
__global__ void k_softmax(float* __restrict__ x, int C, f16* __restrict__ outh) {
  int r = blockIdx.x;
  float* row = x + (size_t)r * C;
  __shared__ float red[256];
  int t = threadIdx.x;
  float mx = -1e30f;
  for (int c = t; c < C; c += 256) mx = fmaxf(mx, row[c]);
  red[t] = mx; __syncthreads();
  for (int s = 128; s > 0; s >>= 1) { if (t < s) red[t] = fmaxf(red[t], red[t + s]); __syncthreads(); }
  mx = red[0]; __syncthreads();
  float sum = 0.f;
  for (int c = t; c < C; c += 256) { float e = __expf(row[c] - mx); row[c] = e; sum += e; }
  red[t] = sum; __syncthreads();
  for (int s = 128; s > 0; s >>= 1) { if (t < s) red[t] += red[t + s]; __syncthreads(); }
  float inv = 1.f / red[0];
  for (int c = t; c < C; c += 256) {
    float v = row[c] * inv;
    row[c] = v;
    if (outh) outh[(size_t)r * C + c] = (f16)v;
  }
}

// ------------------------------------------------------------------ MoE combine
__global__ void k_combine_task(const f16* __restrict__ to, const f16* __restrict__ so,
                               int soRows, int rowOff, const float* __restrict__ sel,
                               f16* __restrict__ out, int Bn, int dout) {
  size_t total = (size_t)Bn * dout;
  size_t stride = (size_t)gridDim.x * blockDim.x;
  for (size_t i = (size_t)blockIdx.x * blockDim.x + threadIdx.x; i < total; i += stride) {
    int b = (int)(i / dout), c = (int)(i % dout);
    const float* s = sel + (size_t)b * NGATE;
    float acc = 0.f;
    #pragma unroll
    for (int e = 0; e < 4; ++e)
      acc += s[e] * (float)to[((size_t)e * Bn + b) * dout + c];
    #pragma unroll
    for (int e = 0; e < 4; ++e)
      acc += s[4 + e] * (float)so[((size_t)e * soRows + rowOff + b) * dout + c];
    out[i] = (f16)acc;
  }
}

__global__ void k_combine_shared(const f16* __restrict__ t0, const f16* __restrict__ t1,
                                 const f16* __restrict__ so, const float* __restrict__ sel,
                                 f16* __restrict__ out, int Bn, int dout) {
  size_t total = (size_t)2 * Bn * dout;
  size_t stride = (size_t)gridDim.x * blockDim.x;
  for (size_t i = (size_t)blockIdx.x * blockDim.x + threadIdx.x; i < total; i += stride) {
    int b = (int)(i / dout), c = (int)(i % dout);
    const f16* ts = (b < Bn) ? (t0 + (size_t)b * dout) : (t1 + (size_t)(b - Bn) * dout);
    const float* s = sel + (size_t)b * NGATE;
    float acc = 0.f;
    #pragma unroll
    for (int e = 0; e < 4; ++e)
      acc += s[e] * (float)ts[(size_t)e * Bn * dout + c];
    #pragma unroll
    for (int e = 0; e < 4; ++e)
      acc += s[4 + e] * (float)so[((size_t)e * 2 * Bn + b) * dout + c];
    out[i] = (f16)acc;
  }
}

// ------------------------------------------------------------------ tower head
__global__ void k_tower_final(const f16* __restrict__ h, const float* __restrict__ W,
                              const float* __restrict__ b, float* __restrict__ out,
                              int Bn, int H) {
  int r = blockIdx.x * blockDim.x + threadIdx.x;
  if (r < Bn) {
    float s = b[0];
    for (int j = 0; j < H; ++j) s += (float)h[(size_t)r * H + j] * W[j];
    out[r] = s;
  }
}

// ================================================================== host side
struct PParams {
  const float *Wq[3], *bq[3], *Wk[3], *bk[3], *Wv[3], *bv[3], *Wo[3], *bo[3];
  const float *gW[3][3], *gb[3][3];
  const float *sW1[3], *sb1[3], *sW2[3], *sb2[3];
  const float *tW1[3][2], *tb1[3][2], *tW2[3][2], *tb2[3][2];
  const float *l1W[2], *l1b[2], *l2W[2], *l2b[2];
  const float *x[3];
};

static const int DIN[3]  = {1024, 1024, 512};
static const int DOUT[3] = {1024, 512, 256};

static void gemm(hipStream_t st, const f16* A, int lda, const f16* Bt, int ldb,
                 const float* bias, float alpha, float* Cf, int ldc, f16* Ch, int ldch,
                 int M, int N, int K, int relu) {
  dim3 g((N + BN_TILE - 1) / BN_TILE, (M + 63) / 64), b(256);
  k_gemm_wmma<<<g, b, 0, st>>>(A, lda, Bt, ldb, bias, alpha, Cf, ldc, Ch, ldch, M, N, K, relu);
}
static void trans(hipStream_t st, const float* src, int R, int C, int ld, f16* dst, int dstld) {
  dim3 g((C + 31) / 32, (R + 31) / 32), b(32, 8);
  k_trans_f32_f16<<<g, b, 0, st>>>(src, R, C, ld, dst, dstld);
}
static void cvt(hipStream_t st, const float* src, f16* dst, long n) {
  int blocks = (int)((n + 255) / 256); if (blocks > 16384) blocks = 16384;
  k_cvt_f32_f16<<<blocks, 256, 0, st>>>(src, dst, n);
}
static void softmax(hipStream_t st, float* x, int R, int C, f16* outh) {
  k_softmax<<<R, 256, 0, st>>>(x, C, outh);
}

extern "C" void kernel_launch(void* const* d_in, const int* in_sizes, int n_in,
                              void* d_out, int out_size, void* d_ws, size_t ws_size,
                              hipStream_t stream) {
  (void)out_size;
  PParams P;

  // ---- input mapping: pytree-sorted, insertion-order, or x's+blob layouts.
  auto fill_sorted = [&](auto get) {   // alphabetical: attn, gates, shared_experts, task_experts, towers
    for (int l = 0; l < 3; ++l) {
      size_t d = DIN[l];
      P.Wk[l] = get(d * d); P.Wo[l] = get(d * d); P.Wq[l] = get(d * d); P.Wv[l] = get(d * d);
      P.bk[l] = get(d);     P.bo[l] = get(d);     P.bq[l] = get(d);     P.bv[l] = get(d);
    }
    for (int l = 0; l < 3; ++l)
      for (int j = 0; j < 3; ++j) { P.gW[l][j] = get((size_t)DIN[l] * NGATE); P.gb[l][j] = get(NGATE); }
    for (int l = 0; l < 3; ++l) {
      P.sW1[l] = get((size_t)4 * DIN[l] * DHID); P.sW2[l] = get((size_t)4 * DHID * DOUT[l]);
      P.sb1[l] = get((size_t)4 * DHID);          P.sb2[l] = get((size_t)4 * DOUT[l]);
    }
    for (int l = 0; l < 3; ++l)
      for (int t = 0; t < 2; ++t) {
        P.tW1[l][t] = get((size_t)4 * DIN[l] * DHID); P.tW2[l][t] = get((size_t)4 * DHID * DOUT[l]);
        P.tb1[l][t] = get((size_t)4 * DHID);          P.tb2[l][t] = get((size_t)4 * DOUT[l]);
      }
    for (int i = 0; i < 2; ++i) {
      P.l1W[i] = get((size_t)DOUT[2] * THID); P.l1b[i] = get(THID);
      P.l2W[i] = get(THID);                   P.l2b[i] = get(1);
    }
  };

  if (n_in >= 89 && in_sizes[0] == BATCH * 1024) {
    // insertion order: x0,x1,x2, shared(W1,b1,W2,b2), task(W1,b1,W2,b2), gates(W,b),
    // attn(Wq,bq,Wk,bk,Wv,bv,Wo,bo), towers(l1W,l1b,l2W,l2b)
    int k = 0;
    auto nxt = [&]() { return (const float*)d_in[k++]; };
    P.x[0] = nxt(); P.x[1] = nxt(); P.x[2] = nxt();
    for (int l = 0; l < 3; ++l) { P.sW1[l]=nxt(); P.sb1[l]=nxt(); P.sW2[l]=nxt(); P.sb2[l]=nxt(); }
    for (int l = 0; l < 3; ++l)
      for (int t = 0; t < 2; ++t) { P.tW1[l][t]=nxt(); P.tb1[l][t]=nxt(); P.tW2[l][t]=nxt(); P.tb2[l][t]=nxt(); }
    for (int l = 0; l < 3; ++l)
      for (int j = 0; j < 3; ++j) { P.gW[l][j]=nxt(); P.gb[l][j]=nxt(); }
    for (int l = 0; l < 3; ++l) {
      P.Wq[l]=nxt(); P.bq[l]=nxt(); P.Wk[l]=nxt(); P.bk[l]=nxt();
      P.Wv[l]=nxt(); P.bv[l]=nxt(); P.Wo[l]=nxt(); P.bo[l]=nxt();
    }
    for (int i = 0; i < 2; ++i) { P.l1W[i]=nxt(); P.l1b[i]=nxt(); P.l2W[i]=nxt(); P.l2b[i]=nxt(); }
  } else if (n_in >= 89) {
    // jax pytree order: params leaves (sorted keys) first, then x0,x1,x2
    int k = 0;
    fill_sorted([&](size_t) { return (const float*)d_in[k++]; });
    P.x[0] = (const float*)d_in[k++]; P.x[1] = (const float*)d_in[k++]; P.x[2] = (const float*)d_in[k++];
  } else {
    // x0,x1,x2 + one flat params blob (sorted-leaf order)
    P.x[0] = (const float*)d_in[0]; P.x[1] = (const float*)d_in[1]; P.x[2] = (const float*)d_in[2];
    const float* blob = (const float*)d_in[3];
    size_t off = 0;
    fill_sorted([&](size_t nelem) { const float* r = blob + off; off += nelem; return r; });
  }

  // ---- workspace bump allocator
  char* wp = (char*)d_ws;
  auto alloc = [&](size_t bytes) { void* r = (void*)wp; wp += (bytes + 255) & ~(size_t)255; return r; };
  f16* hx[2]   = {(f16*)alloc((size_t)BATCH * 1024 * 2), (f16*)alloc((size_t)BATCH * 1024 * 2)};
  f16* hxn[2]  = {(f16*)alloc((size_t)BATCH * 1024 * 2), (f16*)alloc((size_t)BATCH * 1024 * 2)};
  f16* hxs     = (f16*)alloc((size_t)2 * BATCH * 1024 * 2);
  f16* hxsn    = (f16*)alloc((size_t)2 * BATCH * 1024 * 2);
  f16* shin    = (f16*)alloc((size_t)2 * BATCH * 1024 * 2);
  f16* wT      = (f16*)alloc((size_t)DHID * 1024 * 2);
  f16* hh      = (f16*)alloc((size_t)2 * BATCH * DHID * 2);
  f16* taskO[2]= {(f16*)alloc((size_t)4 * BATCH * 1024 * 2), (f16*)alloc((size_t)4 * BATCH * 1024 * 2)};
  f16* sharedO = (f16*)alloc((size_t)4 * 2 * BATCH * 1024 * 2);
  f16* Qb      = (f16*)alloc((size_t)BATCH * 1024 * 2);
  f16* Kb      = (f16*)alloc((size_t)2 * BATCH * 1024 * 2);
  float* Vf    = (float*)alloc((size_t)2 * BATCH * 1024 * 4);
  f16* Vt[2]   = {(f16*)alloc((size_t)512 * 2 * BATCH * 2), (f16*)alloc((size_t)512 * 2 * BATCH * 2)};
  f16* attnO   = (f16*)alloc((size_t)BATCH * 1024 * 2);
  float* scores= (float*)alloc((size_t)BATCH * 2 * BATCH * 4);
  f16* Pb      = (f16*)alloc((size_t)BATCH * 2 * BATCH * 2);
  float* selb  = (float*)alloc((size_t)2 * BATCH * NGATE * 4);
  f16* towerh  = (f16*)alloc((size_t)BATCH * THID * 2);
  if ((size_t)(wp - (char*)d_ws) > ws_size) return;  // workspace too small

  // ---- initial f16 inputs
  cvt(stream, P.x[0], hx[0], (long)BATCH * 1024);
  cvt(stream, P.x[1], hx[1], (long)BATCH * 1024);
  cvt(stream, P.x[2], hxs,  (long)BATCH * 1024);
  int rows_s = BATCH;                 // rows of xs[2]
  const int Rs = 2 * BATCH;           // rows of shared_in / shared experts

  for (int lvl = 0; lvl < 3; ++lvl) {
    const int din = DIN[lvl], dout = DOUT[lvl];
    const int d = din, dh = d / 2, Bk = rows_s;
    const float iscale = 1.f / sqrtf((float)dh);

    // -------- task experts: taskO[i][e] = relu(x_i W1_e + b1_e) W2_e + b2_e
    for (int i = 0; i < 2; ++i)
      for (int e = 0; e < 4; ++e) {
        trans(stream, P.tW1[lvl][i] + (size_t)e * din * DHID, din, DHID, DHID, wT, din);
        gemm(stream, hx[i], din, wT, din, P.tb1[lvl][i] + (size_t)e * DHID, 1.f,
             nullptr, 0, hh, DHID, BATCH, DHID, din, 1);
        trans(stream, P.tW2[lvl][i] + (size_t)e * DHID * dout, DHID, dout, dout, wT, DHID);
        gemm(stream, hh, DHID, wT, DHID, P.tb2[lvl][i] + (size_t)e * dout, 1.f,
             nullptr, 0, taskO[i] + (size_t)e * BATCH * dout, dout, BATCH, dout, DHID, 0);
      }

    // -------- attention (K/V from xs[2]); shin = concat_i oproj(attn_i)
    trans(stream, P.Wk[lvl], d, d, d, wT, d);
    gemm(stream, hxs, d, wT, d, P.bk[lvl], 1.f, nullptr, 0, Kb, d, Bk, d, d, 0);
    trans(stream, P.Wv[lvl], d, d, d, wT, d);
    gemm(stream, hxs, d, wT, d, P.bv[lvl], 1.f, Vf, d, nullptr, 0, Bk, d, d, 0);
    for (int h = 0; h < 2; ++h)                 // per-head V^T (dh x Bk) for P*V
      trans(stream, Vf + h * dh, Bk, dh, d, Vt[h], Bk);
    for (int i = 0; i < 2; ++i) {
      trans(stream, P.Wq[lvl], d, d, d, wT, d);
      gemm(stream, hx[i], din, wT, d, P.bq[lvl], 1.f, nullptr, 0, Qb, d, BATCH, d, d, 0);
      for (int h = 0; h < 2; ++h) {
        gemm(stream, Qb + h * dh, d, Kb + h * dh, d, nullptr, iscale,
             scores, Bk, nullptr, 0, BATCH, Bk, dh, 0);
        softmax(stream, scores, BATCH, Bk, Pb);
        gemm(stream, Pb, Bk, Vt[h], Bk, nullptr, 1.f,
             nullptr, 0, attnO + h * dh, d, BATCH, dh, Bk, 0);
      }
      trans(stream, P.Wo[lvl], d, d, d, wT, d);
      gemm(stream, attnO, d, wT, d, P.bo[lvl], 1.f,
           nullptr, 0, shin + (size_t)i * BATCH * d, d, BATCH, d, d, 0);
    }

    // -------- shared experts on shin (2B rows)
    for (int e = 0; e < 4; ++e) {
      trans(stream, P.sW1[lvl] + (size_t)e * din * DHID, din, DHID, DHID, wT, din);
      gemm(stream, shin, din, wT, din, P.sb1[lvl] + (size_t)e * DHID, 1.f,
           nullptr, 0, hh, DHID, Rs, DHID, din, 1);
      trans(stream, P.sW2[lvl] + (size_t)e * DHID * dout, DHID, dout, dout, wT, DHID);
      gemm(stream, hh, DHID, wT, DHID, P.sb2[lvl] + (size_t)e * dout, 1.f,
           nullptr, 0, sharedO + (size_t)e * Rs * dout, dout, Rs, dout, DHID, 0);
    }

    // -------- gates + combines
    size_t nTask = (size_t)BATCH * dout;
    int cblk = (int)((nTask + 255) / 256); if (cblk > 32768) cblk = 32768;
    for (int i = 0; i < 2; ++i) {
      trans(stream, P.gW[lvl][i], din, NGATE, NGATE, wT, din);
      gemm(stream, hx[i], din, wT, din, P.gb[lvl][i], 1.f,
           selb, NGATE, nullptr, 0, BATCH, NGATE, din, 0);
      softmax(stream, selb, BATCH, NGATE, nullptr);
      k_combine_task<<<cblk, 256, 0, stream>>>(taskO[i], sharedO, Rs, i * BATCH,
                                               selb, hxn[i], BATCH, dout);
    }
    if (lvl < 2) {
      trans(stream, P.gW[lvl][2], din, NGATE, NGATE, wT, din);
      gemm(stream, shin, din, wT, din, P.gb[lvl][2], 1.f,
           selb, NGATE, nullptr, 0, Rs, NGATE, din, 0);
      softmax(stream, selb, Rs, NGATE, nullptr);
      size_t nSh = (size_t)Rs * dout;
      int sblk = (int)((nSh + 255) / 256); if (sblk > 32768) sblk = 32768;
      k_combine_shared<<<sblk, 256, 0, stream>>>(taskO[0], taskO[1], sharedO,
                                                 selb, hxsn, BATCH, dout);
      f16* t = hxs; hxs = hxsn; hxsn = t;
      rows_s = Rs;
    }
    { f16* t = hx[0]; hx[0] = hxn[0]; hxn[0] = t; }
    { f16* t = hx[1]; hx[1] = hxn[1]; hxn[1] = t; }
  }

  // -------- towers: out_i = relu(cgc_i W1 + b1) W2 + b2
  float* out = (float*)d_out;
  for (int i = 0; i < 2; ++i) {
    trans(stream, P.l1W[i], DOUT[2], THID, THID, wT, DOUT[2]);
    gemm(stream, hx[i], DOUT[2], wT, DOUT[2], P.l1b[i], 1.f,
         nullptr, 0, towerh, THID, BATCH, THID, DOUT[2], 1);
    k_tower_final<<<(BATCH + 255) / 256, 256, 0, stream>>>(
        towerh, P.l2W[i], P.l2b[i], out + (size_t)i * BATCH, BATCH, THID);
  }
}
